// Model_35253091565659
// MI455X (gfx1250) — matrix operand order repaired and checked by reading, not compiled, not run on hardware
//
#include <hip/hip_runtime.h>
#include <hip/hip_bf16.h>

typedef __attribute__((ext_vector_type(16))) __bf16 v16bf;
typedef __attribute__((ext_vector_type(8)))  __bf16 v8bf;
typedef __attribute__((ext_vector_type(4)))  __bf16 v4bf;
typedef __attribute__((ext_vector_type(8)))  float  v8f;

#define BM 128
#define BN 64
#define BK 32

__device__ __forceinline__ v16bf mk16(const __bf16* p0, const __bf16* p1) {
  v8bf lo = *(const v8bf*)p0;
  v8bf hi = *(const v8bf*)p1;
  return __builtin_shufflevector(lo, hi, 0,1,2,3,4,5,6,7,8,9,10,11,12,13,14,15);
}

__device__ __forceinline__ void wait_async0() {
#if __has_builtin(__builtin_amdgcn_s_wait_asynccnt)
  __builtin_amdgcn_s_wait_asynccnt(0);
#else
  asm volatile("s_wait_asynccnt 0x0" ::: "memory");
#endif
}

// ---------------------------------------------------------------------------
// C[M,N] = act(A[M,K] @ B[Npad,K]^T + bias), bf16 operands, f32 accumulate.
// Preconditions (guaranteed by padded operand materialization):
//   K % 32 == 0, M % 128 == 0, grid.x * 64 == Npad (B has Npad zero-padded rows)
// 256 threads = 8 waves; block tile 128x64; wave owns 32x32 (4 WMMA / chunk).
// Double-buffered LDS; A tile via global_load_async_to_lds_b128 pipelined one
// chunk ahead (ASYNCcnt); B tile copied n-major through VGPRs; 1 barrier / chunk.
// ---------------------------------------------------------------------------
__global__ __launch_bounds__(256) void wmma_gemm_bf16(
    const __bf16* __restrict__ A, const __bf16* __restrict__ B,
    const float* __restrict__ bias, float* __restrict__ Cf,
    __bf16* __restrict__ Cb, int M, int N, int K, int ldc, int act)
{
  __shared__ __align__(16) __bf16 sA[2][BM * BK];   // 2 x 8 KB
  __shared__ __align__(16) __bf16 sB[2][BK * BN];   // 2 x 4 KB (n-major)

  const int tid  = threadIdx.x;
  const int lane = tid & 31;
  const int w    = tid >> 5;
  const int sm   = (w >> 1) * 32;
  const int sn   = (w & 1) * 32;
  const int m0   = blockIdx.y * BM;
  const int n0   = blockIdx.x * BN;

  const unsigned long long Abase = (unsigned long long)(uintptr_t)A;

  // per-thread staging coordinates
  const int ar0 = (tid + 0)   >> 2, ako0 = ((tid + 0)   & 3) * 8;  // A chunk 0
  const int ar1 = (tid + 256) >> 2, ako1 = ((tid + 256) & 3) * 8;  // A chunk 1
  const int bn_ = tid >> 2,         bko  = (tid & 3) * 8;          // B chunk

  auto stageA = [&](int buf, int k0) {
    unsigned lds0 = (unsigned)(uintptr_t)(&sA[buf][ar0 * BK + ako0]);
    unsigned off0 = (unsigned)(((size_t)(m0 + ar0) * K + (k0 + ako0)) * sizeof(__bf16));
    asm volatile("global_load_async_to_lds_b128 %0, %1, %2"
                 :: "v"(lds0), "v"(off0), "s"(Abase) : "memory");
    unsigned lds1 = (unsigned)(uintptr_t)(&sA[buf][ar1 * BK + ako1]);
    unsigned off1 = (unsigned)(((size_t)(m0 + ar1) * K + (k0 + ako1)) * sizeof(__bf16));
    asm volatile("global_load_async_to_lds_b128 %0, %1, %2"
                 :: "v"(lds1), "v"(off1), "s"(Abase) : "memory");
  };
  auto loadB = [&](int k0) -> v8bf {
    return *(const v8bf*)(B + (size_t)(n0 + bn_) * K + k0 + bko);
  };
  auto storeB = [&](int buf, v8bf bv) {
#pragma unroll
    for (int i = 0; i < 8; ++i) sB[buf][bn_ * BK + bko + i] = bv[i];
  };

  v8f acc[2][2] = {};
  const int nk = K / BK;

  // prologue: stage chunk 0 into buffer 0
  stageA(0, 0);
  storeB(0, loadB(0));
  wait_async0();
  __syncthreads();

  for (int kt = 0; kt < nk; ++kt) {
    const int cur = kt & 1;
    const int nxt = cur ^ 1;
    const bool hasNext = (kt + 1) < nk;

    // issue next-chunk copies first: they run under this chunk's WMMAs
    v8bf bvn;
    if (hasNext) {
      stageA(nxt, (kt + 1) * BK);
      bvn = loadB((kt + 1) * BK);
    }

    // fragments for current chunk (ISA 7.12.2 layouts, ds_load_b128)
    const int arow = lane & 15;
    const int akb  = (lane >> 4) * 8;
    const __bf16* pa0 = &sA[cur][(sm + arow) * BK];
    const __bf16* pa1 = &sA[cur][(sm + 16 + arow) * BK];
    v16bf a0 = mk16(pa0 + akb, pa0 + akb + 16);
    v16bf a1 = mk16(pa1 + akb, pa1 + akb + 16);
    const __bf16* pb = &sB[cur][(sn + arow) * BK];     // lane holds B-column N=lane&15
    v16bf b0 = mk16(pb + akb, pb + akb + 16);
    v16bf b1 = mk16(pb + 16 * BK + akb, pb + 16 * BK + akb + 16);

    acc[0][0] = __builtin_amdgcn_wmma_f32_16x16x32_bf16(false, a0, false, b0, (short)0, acc[0][0], false, false);
    acc[0][1] = __builtin_amdgcn_wmma_f32_16x16x32_bf16(false, a0, false, b1, (short)0, acc[0][1], false, false);
    acc[1][0] = __builtin_amdgcn_wmma_f32_16x16x32_bf16(false, a1, false, b0, (short)0, acc[1][0], false, false);
    acc[1][1] = __builtin_amdgcn_wmma_f32_16x16x32_bf16(false, a1, false, b1, (short)0, acc[1][1], false, false);

    if (hasNext) {
      storeB(nxt, bvn);      // disjoint buffer: safe while others read cur
      wait_async0();         // drain next-chunk A copies issued above
    }
    __syncthreads();
  }

  // ---- epilogue: row = base + r + 8*(lane>>4); col = base + (lane&15) ----
#pragma unroll
  for (int i = 0; i < 2; ++i) {
#pragma unroll
    for (int j = 0; j < 2; ++j) {
      int col = n0 + sn + 16 * j + (lane & 15);
      int rb  = m0 + sm + 16 * i + ((lane >> 4) * 8);
      float bb = (bias != nullptr && col < N) ? bias[col] : 0.0f;
#pragma unroll
      for (int r = 0; r < 8; ++r) {
        int row = rb + r;
        if (row < M && col < N) {
          float v = acc[i][j][r] + bb;
          if (act == 1) v = (v > 20.0f) ? v : log1pf(expf(v));  // softplus
          if (Cf) Cf[(size_t)row * ldc + col] = v;
          if (Cb) Cb[(size_t)row * ldc + col] = (__bf16)v;
        }
      }
    }
  }
}

// f32 -> bf16 with 2D zero-padding: dst[rowsPad x Kp], src[rows x K]
// grid: ((Kp/4 + 255)/256, rowsPad)
__global__ void cvt_pad_bf16_kernel(const float* __restrict__ src,
                                    __bf16* __restrict__ dst,
                                    int rows, int K, int Kp)
{
  int r = blockIdx.y;
  int k = (blockIdx.x * blockDim.x + threadIdx.x) * 4;
  if (k >= Kp) return;
  v4bf b;
  if (r < rows && k + 4 <= K) {
    float4 f = *(const float4*)(src + (size_t)r * K + k);
    b[0] = (__bf16)f.x; b[1] = (__bf16)f.y;
    b[2] = (__bf16)f.z; b[3] = (__bf16)f.w;
  } else {
#pragma unroll
    for (int i = 0; i < 4; ++i) {
      int kk = k + i;
      float v = (r < rows && kk < K) ? src[(size_t)r * K + kk] : 0.0f;
      b[i] = (__bf16)v;
    }
  }
  *(v4bf*)(dst + (size_t)r * Kp + k) = b;
}

// ---------------------------------------------------------------------------
__global__ __launch_bounds__(256) void bn_stats_kernel(
    const float* __restrict__ mv, const float* __restrict__ gm,
    const float* __restrict__ betam, const float* __restrict__ gv,
    const float* __restrict__ betav, float* __restrict__ stats)
{
  const int c = blockIdx.x;
  __shared__ float s1[256], s2[256];
  float a = 0.0f, b = 0.0f;
  for (int n = threadIdx.x; n < 2048; n += 256) {
    float x = mv[(size_t)n * 128 + c];
    a += x; b += x * x;
  }
  s1[threadIdx.x] = a; s2[threadIdx.x] = b;
  __syncthreads();
  for (int s = 128; s > 0; s >>= 1) {
    if (threadIdx.x < s) {
      s1[threadIdx.x] += s1[threadIdx.x + s];
      s2[threadIdx.x] += s2[threadIdx.x + s];
    }
    __syncthreads();
  }
  if (threadIdx.x == 0) {
    float mean = s1[0] * (1.0f / 2048.0f);
    float var  = s2[0] * (1.0f / 2048.0f) - mean * mean;
    float g  = (c < 64) ? gm[c] : gv[c - 64];
    float be = (c < 64) ? betam[c] : betav[c - 64];
    float sc = g * rsqrtf(var + 1e-5f);
    stats[c]       = sc;
    stats[128 + c] = be - mean * sc;
  }
}

__global__ void z_kernel(const float* __restrict__ mv, const float* __restrict__ stats,
                         const float* __restrict__ eps, float* __restrict__ z_out,
                         float* __restrict__ zx_out)
{
  int i = blockIdx.x * blockDim.x + threadIdx.x;
  if (i >= 2048 * 64) return;
  int n = i >> 6, c = i & 63;
  float m  = mv[(size_t)n * 128 + c]      * stats[c]      + stats[128 + c];
  float lv = mv[(size_t)n * 128 + 64 + c] * stats[64 + c] + stats[192 + c];
  float zv = m + sqrtf(expf(lv)) * eps[i];
  z_out[i]  = zv;
  zx_out[i] = zv;
}

__global__ void copy_kernel(float* __restrict__ dst, const float* __restrict__ src, int n)
{
  int i = blockIdx.x * blockDim.x + threadIdx.x;
  if (i < n) dst[i] = src[i];
}

// RBF distances + row softmax: one block per sample; writes f32 + bf16 phi.
__global__ __launch_bounds__(256) void dist_softmax_kernel(
    const float* __restrict__ z, const float* __restrict__ centres,
    float* __restrict__ phi, __bf16* __restrict__ phi_bf)
{
  const int n = blockIdx.x;
  __shared__ float zr[64];
  __shared__ float red[256];
  if (threadIdx.x < 64) zr[threadIdx.x] = z[(size_t)n * 64 + threadIdx.x];
  __syncthreads();

  float v0, v1;
  {
    const float* c0 = centres + (size_t)threadIdx.x * 64;
    const float* c1 = centres + (size_t)(threadIdx.x + 256) * 64;
    float d0 = 0.0f, d1 = 0.0f;
#pragma unroll 8
    for (int k = 0; k < 64; ++k) {
      float a = zr[k] - c0[k]; d0 += a * a;
      float b = zr[k] - c1[k]; d1 += b * b;
    }
    v0 = -0.5f * d0; v1 = -0.5f * d1;
  }
  red[threadIdx.x] = fmaxf(v0, v1);
  __syncthreads();
  for (int s = 128; s > 0; s >>= 1) {
    if (threadIdx.x < s) red[threadIdx.x] = fmaxf(red[threadIdx.x], red[threadIdx.x + s]);
    __syncthreads();
  }
  float mx = red[0];
  __syncthreads();
  float e0 = expf(v0 - mx), e1 = expf(v1 - mx);
  red[threadIdx.x] = e0 + e1;
  __syncthreads();
  for (int s = 128; s > 0; s >>= 1) {
    if (threadIdx.x < s) red[threadIdx.x] += red[threadIdx.x + s];
    __syncthreads();
  }
  float inv = 1.0f / red[0];
  float p0 = e0 * inv, p1 = e1 * inv;
  phi[(size_t)n * 512 + threadIdx.x]          = p0;
  phi[(size_t)n * 512 + threadIdx.x + 256]    = p1;
  phi_bf[(size_t)n * 512 + threadIdx.x]       = (__bf16)p0;
  phi_bf[(size_t)n * 512 + threadIdx.x + 256] = (__bf16)p1;
}

// Row softmax over V=30000 -> betaT[v*512 + t] (bf16), zero-fills pad rows.
__global__ __launch_bounds__(256) void softmax_beta_kernel(
    const float* __restrict__ logits, __bf16* __restrict__ betaT)
{
  const int t = blockIdx.x;
  const float* row = logits + (size_t)t * 30000;
  __shared__ float red[256];
  float mx = -1e30f;
  for (int v = threadIdx.x; v < 30000; v += 256) mx = fmaxf(mx, row[v]);
  red[threadIdx.x] = mx;
  __syncthreads();
  for (int s = 128; s > 0; s >>= 1) {
    if (threadIdx.x < s) red[threadIdx.x] = fmaxf(red[threadIdx.x], red[threadIdx.x + s]);
    __syncthreads();
  }
  mx = red[0];
  __syncthreads();
  float sum = 0.0f;
  for (int v = threadIdx.x; v < 30000; v += 256) sum += expf(row[v] - mx);
  red[threadIdx.x] = sum;
  __syncthreads();
  for (int s = 128; s > 0; s >>= 1) {
    if (threadIdx.x < s) red[threadIdx.x] += red[threadIdx.x + s];
    __syncthreads();
  }
  float inv = 1.0f / red[0];
  for (int v = threadIdx.x; v < 30000; v += 256)
    betaT[(size_t)v * 512 + t] = (__bf16)(expf(row[v] - mx) * inv);
  if (threadIdx.x < 16)   // zero pad rows 30000..30015 (Npad = 30016)
    betaT[(size_t)(30000 + threadIdx.x) * 512 + t] = (__bf16)0.0f;
}

// ---------------------------------------------------------------------------
extern "C" void kernel_launch(void* const* d_in, const int* in_sizes, int n_in,
                              void* d_out, int out_size, void* d_ws, size_t ws_size,
                              hipStream_t stream)
{
  const float* input   = (const float*)d_in[0];
  const float* eps     = (const float*)d_in[2];
  const float* W1      = (const float*)d_in[3];
  const float* b1      = (const float*)d_in[4];
  const float* W2      = (const float*)d_in[5];
  const float* b2      = (const float*)d_in[6];
  const float* Wm      = (const float*)d_in[7];
  const float* bm      = (const float*)d_in[8];
  const float* Wv      = (const float*)d_in[9];
  const float* bv      = (const float*)d_in[10];
  const float* gm      = (const float*)d_in[11];
  const float* betam   = (const float*)d_in[12];
  const float* gv      = (const float*)d_in[13];
  const float* betav   = (const float*)d_in[14];
  const float* centres = (const float*)d_in[15];
  const float* mu_z    = (const float*)d_in[16];
  const float* emb     = (const float*)d_in[17];

  // output slots (return order: z, recon_v, zx, zc, zx_phi)
  float* out = (float*)d_out;
  float* o_z     = out;
  float* o_recon = o_z + 2048 * 64;
  float* o_zx    = o_recon + (size_t)2048 * 30000;
  float* o_zc    = o_zx + 2048 * 64;
  float* o_phi   = o_zc + 512 * 64;

  // padded dims: K -> mult of 32, B-rows -> mult of 64
  const int VKP = 30016;   // 30000 padded (K of en1 GEMM; also Npad of V-sized B)
  const int EKP = 320;     // 300 padded

  // workspace carve-out
  char* wp = (char*)d_ws;
  auto carve = [&](size_t bytes) -> void* {
    void* p = wp;
    wp += (bytes + 255) & ~(size_t)255;
    return p;
  };
  __bf16* input_bf = (__bf16*)carve((size_t)2048 * VKP * 2);
  __bf16* W1_bf    = (__bf16*)carve((size_t)1024 * VKP * 2);
  __bf16* W2_bf    = (__bf16*)carve((size_t)1024 * 1024 * 2);
  __bf16* Wm_bf    = (__bf16*)carve((size_t)64 * 1024 * 2);
  __bf16* Wv_bf    = (__bf16*)carve((size_t)64 * 1024 * 2);
  __bf16* muz_bf   = (__bf16*)carve((size_t)512 * EKP * 2);
  __bf16* emb_bf   = (__bf16*)carve((size_t)VKP * EKP * 2);
  __bf16* en1_bf   = (__bf16*)carve((size_t)2048 * 1024 * 2);
  __bf16* en2_bf   = (__bf16*)carve((size_t)2048 * 1024 * 2);
  __bf16* phi_bf   = (__bf16*)carve((size_t)2048 * 512 * 2);
  __bf16* betaT_bf = (__bf16*)carve((size_t)VKP * 512 * 2);
  float*  mv       = (float*)carve((size_t)2048 * 128 * 4);
  float*  stats    = (float*)carve(256 * 4);
  float*  logits   = (float*)carve((size_t)512 * 30000 * 4);

  dim3 blk(256);
  auto cvt = [&](const float* s, __bf16* d, int rows, int rowsPad, int K, int Kp) {
    dim3 grid((Kp / 4 + 255) / 256, rowsPad);
    hipLaunchKernelGGL(cvt_pad_bf16_kernel, grid, blk, 0, stream, s, d, rows, K, Kp);
  };
  auto gemm = [&](const __bf16* A, const __bf16* B, const float* bias,
                  float* Cf, __bf16* Cb, int M, int N, int Npad, int K, int ldc, int act) {
    dim3 grid(Npad / BN, M / BM);
    hipLaunchKernelGGL(wmma_gemm_bf16, grid, blk, 0, stream, A, B, bias, Cf, Cb, M, N, K, ldc, act);
  };

  // operand conversions with zero padding (bandwidth-trivial, done once)
  cvt(input, input_bf, 2048,  2048,  30000, VKP);
  cvt(W1,    W1_bf,    1024,  1024,  30000, VKP);
  cvt(W2,    W2_bf,    1024,  1024,  1024,  1024);
  cvt(Wm,    Wm_bf,    64,    64,    1024,  1024);
  cvt(Wv,    Wv_bf,    64,    64,    1024,  1024);
  cvt(mu_z,  muz_bf,   512,   512,   300,   EKP);
  cvt(emb,   emb_bf,   30000, VKP,   300,   EKP);

  // encoder
  gemm(input_bf, W1_bf, b1, nullptr, en1_bf, 2048, 1024, 1024, VKP,  1024, 1);
  gemm(en1_bf,   W2_bf, b2, nullptr, en2_bf, 2048, 1024, 1024, 1024, 1024, 1);
  gemm(en2_bf,   Wm_bf, bm, mv,      nullptr, 2048, 64, 64, 1024, 128, 0);
  gemm(en2_bf,   Wv_bf, bv, mv + 64, nullptr, 2048, 64, 64, 1024, 128, 0);

  hipLaunchKernelGGL(bn_stats_kernel, dim3(128), blk, 0, stream, mv, gm, betam, gv, betav, stats);
  hipLaunchKernelGGL(z_kernel, dim3((2048 * 64 + 255) / 256), blk, 0, stream,
                     mv, stats, eps, o_z, o_zx);
  hipLaunchKernelGGL(copy_kernel, dim3((512 * 64 + 255) / 256), blk, 0, stream,
                     o_zc, centres, 512 * 64);
  hipLaunchKernelGGL(dist_softmax_kernel, dim3(2048), blk, 0, stream,
                     o_z, centres, o_phi, phi_bf);

  // decoder
  gemm(muz_bf, emb_bf, nullptr, logits, nullptr, 512, 30000, VKP, EKP, 30000, 0);
  hipLaunchKernelGGL(softmax_beta_kernel, dim3(512), blk, 0, stream, logits, betaT_bf);
  gemm(phi_bf, betaT_bf, nullptr, o_recon, nullptr, 2048, 30000, VKP, 512, 30000, 0);
}
